// TopKGatingRouter_68899865362460
// MI455X (gfx1250) — matrix operand order repaired and checked by reading, not compiled, not run on hardware
//
#include <hip/hip_runtime.h>
#include <hip/hip_bf16.h>

// ---------------------------------------------------------------------------
// MoE top-k gating router, fused for MI455X (gfx1250, wave32, WMMA).
//   gate_logits = x @ W^T   via v_wmma_f32_16x16x32_bf16 (fp32 accum)
//   softmax + top2 + normalize + aux-loss reductions + expert histogram fused.
// x is 268 MB read once -> ~11.5 us floor @ 23.3 TB/s. W (1 MB) is converted
// to bf16 once in workspace, then double-buffered into LDS with
// GLOBAL_LOAD_ASYNC_TO_LDS_B128 (ASYNCcnt) so the hot loop is loads + cvt
// + WMMA only.
// ---------------------------------------------------------------------------

typedef __attribute__((ext_vector_type(16))) __bf16          v16bf;
typedef __attribute__((ext_vector_type(8)))  __bf16          v8bf;
typedef __attribute__((ext_vector_type(16))) unsigned short  v16u;
typedef __attribute__((ext_vector_type(8)))  unsigned short  v8u;
typedef __attribute__((ext_vector_type(4)))  float           v4f;
typedef __attribute__((ext_vector_type(8)))  float           v8f;
typedef __attribute__((ext_vector_type(16))) float           v16f;
typedef __attribute__((ext_vector_type(4)))  int             v4i;

#define H_DIM     4096
#define EXPERTS   64
#define NTOK      16384
#define TOK_BLK   128           // tokens per workgroup (8 waves x 16 tokens)
#define NCHUNK    8
#define CHUNK     512           // K-chunk staged in LDS per iteration
#define WROW      (CHUNK + 8)   // padded LDS row stride (bf16 elems)
#define RW_BASE   0
#define IDX_BASE  (NTOK * 2)            // 32768
#define AUX_BASE  (NTOK * 4)            // 65536
#define CNT_BASE  (NTOK * 4 + 1)        // 65537
#define WS_RED    512                    // bytes reserved for reductions
#define WS_WBF    (EXPERTS * H_DIM * 2)  // bf16 W copy

#if __has_builtin(__builtin_amdgcn_global_load_async_to_lds_b128) && \
    __has_builtin(__builtin_amdgcn_s_wait_asynccnt)
#define HAVE_ASYNC_LDS 1
#else
#define HAVE_ASYNC_LDS 0
#endif

struct SMem {
  union {
    unsigned short w[2][EXPERTS * WROW];       // 2 x 66,560 B bf16 W chunks
    struct {
      float logits[TOK_BLK * EXPERTS];         // 32 KB
      float mx[TOK_BLK];
      float rinv[TOK_BLK];
      float cnt[EXPERTS];
    } ep;
  };
};

// Vectorized fp32 -> bf16 (RNE); backend picks packed cvt if available.
__device__ __forceinline__ v8bf cvt8(v4f a, v4f b) {
  v8f f = __builtin_shufflevector(a, b, 0, 1, 2, 3, 4, 5, 6, 7);
  return __builtin_convertvector(f, v8bf);
}

__device__ __forceinline__ v16bf cvt16(v4f f0, v4f f1, v4f f2, v4f f3) {
  v8f lo = __builtin_shufflevector(f0, f1, 0, 1, 2, 3, 4, 5, 6, 7);
  v8f hi = __builtin_shufflevector(f2, f3, 0, 1, 2, 3, 4, 5, 6, 7);
  v16f f = __builtin_shufflevector(lo, hi, 0, 1, 2, 3, 4, 5, 6, 7,
                                           8, 9, 10, 11, 12, 13, 14, 15);
  return __builtin_convertvector(f, v16bf);
}

__device__ __forceinline__ v8f wmma_bf16(v16bf a, v16bf b, v8f c) {
  return __builtin_amdgcn_wmma_f32_16x16x32_bf16(false, a, false, b, (short)0, c,
                                                 false, false);
}

__device__ __forceinline__ v16bf lds_b16(const unsigned short* p) {
  v8u b0 = *(const v8u*)p;
  v8u b1 = *(const v8u*)(p + 8);
  v16u u = __builtin_shufflevector(b0, b1, 0, 1, 2, 3, 4, 5, 6, 7,
                                          8, 9, 10, 11, 12, 13, 14, 15);
  return __builtin_bit_cast(v16bf, u);
}

// Cooperative stage of one bf16 W chunk (64 x 512) into padded LDS tile.
__device__ __forceinline__ void stage_w(const unsigned short* __restrict__ Wbf,
                                        unsigned short* dst_base, int kk, int tid) {
  const int row = tid >> 2;          // expert row, 4 threads per row
  const int q   = tid & 3;
  const unsigned short* src = Wbf + (size_t)row * H_DIM + kk + q * 128;
  unsigned short* dst = dst_base + row * WROW + q * 128;
#if HAVE_ASYNC_LDS
#pragma unroll
  for (int i = 0; i < 16; ++i)        // 16 x b128 async copies per thread
    __builtin_amdgcn_global_load_async_to_lds_b128(
        (v4i*)(src + i * 8), (v4i*)(dst + i * 8), 0, 0);
#else
#pragma unroll
  for (int i = 0; i < 16; ++i)
    *(v8u*)(dst + i * 8) = *(const v8u*)(src + i * 8);
#endif
}

__device__ __forceinline__ void wait_async_lds() {
#if HAVE_ASYNC_LDS
  __builtin_amdgcn_s_wait_asynccnt(0);
#endif
}

// ---------------------------------------------------------------------------
__global__ __launch_bounds__(256) void router_init(float* g_red) {
  if (threadIdx.x < 128) g_red[threadIdx.x] = 0.0f;   // probsum[64] + cnt[64]
}

// One-shot W fp32 -> bf16 (1 MB -> 512 KB). 128 blocks x 256 thr x 8 elems.
__global__ __launch_bounds__(256) void w_convert(const float* __restrict__ W,
                                                 unsigned short* __restrict__ Wbf) {
  const int i = (blockIdx.x * 256 + threadIdx.x) * 8;
  v4f a = *(const v4f*)(W + i);
  v4f b = *(const v4f*)(W + i + 4);
  v8bf r = cvt8(a, b);
  *(v8u*)(Wbf + i) = __builtin_bit_cast(v8u, r);
}

// ---------------------------------------------------------------------------
template <bool USE_WBF>
__global__ __launch_bounds__(256) void router_main(const float* __restrict__ x,
                                                   const float* __restrict__ W,
                                                   const unsigned short* __restrict__ Wbf,
                                                   float* __restrict__ out,
                                                   float* __restrict__ g_probsum,
                                                   float* __restrict__ g_cnt) {
  __shared__ SMem sm;
  const int tid  = threadIdx.x;
  const int wave = tid >> 5;
  const int lane = tid & 31;
  const int l16  = lane & 15;
  const int hi   = lane >> 4;          // 0: low K half, 1: high K half
  const int blk  = blockIdx.x;
  const int t0   = blk * TOK_BLK + wave * 16;

  v8f acc0 = {}, acc1 = {}, acc2 = {}, acc3 = {};

  // A-side: this lane's x row + K-half base (contiguous 16-K halves, matches B).
  const float* xrow = x + (size_t)(t0 + l16) * H_DIM + hi * 16;

  if (USE_WBF) stage_w(Wbf, sm.w[0], 0, tid);   // prime the pipeline

  for (int c = 0; c < NCHUNK; ++c) {
    const int kk = c * CHUNK;

    if (USE_WBF) {
      wait_async_lds();                 // this wave's copies into buf[c&1] done
      __syncthreads();                  // all waves' copies done + prev compute done
      if (c + 1 < NCHUNK)               // prefetch next chunk into other buffer
        stage_w(Wbf, sm.w[(c + 1) & 1], kk + CHUNK, tid);
    } else {
      // Fallback: convert fp32 W chunk in-kernel (small ws_size).
      __syncthreads();
      const int row = tid >> 2, q = tid & 3;
      const float* wsrc = W + (size_t)row * H_DIM + kk + q * 128;
      unsigned short* wdst = sm.w[0] + row * WROW + q * 128;
#pragma unroll
      for (int i = 0; i < 128; i += 8) {
        v4f a = *(const v4f*)(wsrc + i);
        v4f b = *(const v4f*)(wsrc + i + 4);
        v8bf r = cvt8(a, b);
        *(v8u*)(wdst + i) = __builtin_bit_cast(v8u, r);
      }
      __syncthreads();
    }

    const unsigned short* wb = USE_WBF ? sm.w[c & 1] : sm.w[0];
    __builtin_prefetch(xrow + kk + CHUNK, 0, 1);   // global_prefetch_b8: next x chunk

#pragma unroll 4
    for (int s = 0; s < CHUNK / 32; ++s) {
      const int k2 = kk + s * 32;
      const int koff = s * 32 + hi * 16;

      // Issue all four B-tile loads first (distinct regs -> 8x ds_load_b128
      // clause, staggered dscnt waits) so LDS latency overlaps A cvt below.
      v16bf b0 = lds_b16(wb + (0 * 16 + l16) * WROW + koff);
      v16bf b1 = lds_b16(wb + (1 * 16 + l16) * WROW + koff);
      v16bf b2 = lds_b16(wb + (2 * 16 + l16) * WROW + koff);
      v16bf b3 = lds_b16(wb + (3 * 16 + l16) * WROW + koff);

      // A tile 16x32: k {0..15} (lanes<16) / {16..31} (>=16)
      v4f f0 = *(const v4f*)(xrow + k2);
      v4f f1 = *(const v4f*)(xrow + k2 + 4);
      v4f f2 = *(const v4f*)(xrow + k2 + 8);
      v4f f3 = *(const v4f*)(xrow + k2 + 12);
      v16bf av = cvt16(f0, f1, f2, f3);

      acc0 = wmma_bf16(av, b0, acc0);
      acc1 = wmma_bf16(av, b1, acc1);
      acc2 = wmma_bf16(av, b2, acc2);
      acc3 = wmma_bf16(av, b3, acc3);
    }
  }

  wait_async_lds();
  __syncthreads();   // all waves done with W tiles; safe to overlay epilogue

  // Spill accumulators: C/D layout -> lane l16 = expert col, VGPR r = token r+8*hi.
  {
    float* lg = sm.ep.logits + (wave * 16) * EXPERTS;
#pragma unroll
    for (int r = 0; r < 8; ++r) {
      const int trow = r + 8 * hi;
      lg[trow * EXPERTS + 0 * 16 + l16] = acc0[r];
      lg[trow * EXPERTS + 1 * 16 + l16] = acc1[r];
      lg[trow * EXPERTS + 2 * 16 + l16] = acc2[r];
      lg[trow * EXPERTS + 3 * 16 + l16] = acc3[r];
    }
  }
  if (tid < EXPERTS) sm.ep.cnt[tid] = 0.0f;
  __syncthreads();

  // Per-token softmax stats + top-2 (strict '>' keeps lowest index on ties,
  // matching jax.lax.top_k).
  if (tid < TOK_BLK) {
    const float* L = sm.ep.logits + tid * EXPERTS;
    float m = L[0];
#pragma unroll
    for (int e = 1; e < EXPERTS; ++e) m = fmaxf(m, L[e]);
    float sum = 0.0f, p1 = -1.0f, p2 = -1.0f;
    int i1 = 0, i2 = 0;
#pragma unroll
    for (int e = 0; e < EXPERTS; ++e) {
      float p = __expf(L[e] - m);
      sum += p;
      if (p > p1)      { p2 = p1; i2 = i1; p1 = p; i1 = e; }
      else if (p > p2) { p2 = p;  i2 = e; }
    }
    const int g = blk * TOK_BLK + tid;
    const float rn = 1.0f / (p1 + p2);            // softmax scale cancels
    out[RW_BASE  + g * 2 + 0] = p1 * rn;
    out[RW_BASE  + g * 2 + 1] = p2 * rn;
    out[IDX_BASE + g * 2 + 0] = (float)i1;
    out[IDX_BASE + g * 2 + 1] = (float)i2;
    sm.ep.mx[tid]   = m;
    sm.ep.rinv[tid] = 1.0f / sum;
    atomicAdd(&sm.ep.cnt[i1], 1.0f);              // ds_add_f32
    atomicAdd(&sm.ep.cnt[i2], 1.0f);
  }
  __syncthreads();

  // Column reduction of softmax probs (lane == bank, conflict-free) + globals.
  if (tid < EXPERTS) {
    float ps = 0.0f;
#pragma unroll 8
    for (int t = 0; t < TOK_BLK; ++t)
      ps += __expf(sm.ep.logits[t * EXPERTS + tid] - sm.ep.mx[t]) * sm.ep.rinv[t];
    atomicAdd(&g_probsum[tid], ps);               // global_atomic_add_f32
    atomicAdd(&g_cnt[tid], sm.ep.cnt[tid]);
  }
}

// ---------------------------------------------------------------------------
__global__ __launch_bounds__(64) void router_finalize(const float* __restrict__ g_probsum,
                                                      const float* __restrict__ g_cnt,
                                                      float* __restrict__ out) {
  __shared__ float sp[EXPERTS], sa[EXPERTS];
  const int e = threadIdx.x;
  const float inv_n = 1.0f / (float)NTOK;
  float p = g_probsum[e] * inv_n;
  float a = g_cnt[e] * inv_n;
  sp[e] = p * p;
  sa[e] = a * a;
  out[CNT_BASE + e] = g_cnt[e];
  __syncthreads();
  if (e == 0) {
    float s1 = 0.0f, s2 = 0.0f;
#pragma unroll
    for (int i = 0; i < EXPERTS; ++i) { s1 += sp[i]; s2 += sa[i]; }
    out[AUX_BASE] = 0.01f * (s1 * (float)EXPERTS + s2 * (float)EXPERTS);
  }
}

// ---------------------------------------------------------------------------
extern "C" void kernel_launch(void* const* d_in, const int* in_sizes, int n_in,
                              void* d_out, int out_size, void* d_ws, size_t ws_size,
                              hipStream_t stream) {
  const float* x = (const float*)d_in[0];   // [4,4096,4096] fp32
  const float* W = (const float*)d_in[1];   // [64,4096]     fp32
  float* out = (float*)d_out;               // [rw 32768 | idx 32768 | aux 1 | cnt 64]
  float* g_probsum = (float*)d_ws;          // [64]
  float* g_cnt     = g_probsum + EXPERTS;   // [64]
  unsigned short* Wbf = (unsigned short*)((char*)d_ws + WS_RED);

  const bool use_wbf = ws_size >= (size_t)(WS_RED + WS_WBF);

  router_init<<<1, 256, 0, stream>>>(g_probsum);
  if (use_wbf) {
    w_convert<<<128, 256, 0, stream>>>(W, Wbf);
    router_main<true><<<NTOK / TOK_BLK, 256, 0, stream>>>(x, W, Wbf, out,
                                                          g_probsum, g_cnt);
  } else {
    router_main<false><<<NTOK / TOK_BLK, 256, 0, stream>>>(x, W, Wbf, out,
                                                           g_probsum, g_cnt);
  }
  router_finalize<<<1, 64, 0, stream>>>(g_probsum, g_cnt, out);
}
